// BatchMultiHeadGraphAttention_69277822485309
// MI455X (gfx1250) — matrix operand order repaired and checked by reading, not compiled, not run on hardware
//
#include <hip/hip_runtime.h>

// ---- MI455X / gfx1250 fused GAT ----
// Phase 1: h_prime = h@w via v_wmma_f32_16x16x32_f16 (w staged transposed in LDS),
//          tanh + a_src/a_dst dots reduced cross-lane, h_prime stored transposed f16.
// Phase 2: flash-attention over j: masked leaky scores -> online softmax ->
//          P(16x32,f16) @ V(32x16,f16) WMMA accumulate in f32. attn never hits HBM.
//          global_prefetch_b8 issued one j-block ahead for adj and V streams.

typedef __attribute__((ext_vector_type(16))) _Float16 v16h;
typedef __attribute__((ext_vector_type(8)))  _Float16 v8h;
typedef __attribute__((ext_vector_type(8)))  float    v8f;
typedef __attribute__((ext_vector_type(4)))  float    v4f;

#define BB 4
#define CC 4
#define HHD 4
#define NN 1024
#define FF 64
#define BCH (BB*CC*HHD)   // 64
#define NEGINF (-1e12f)
#define SLOPE 0.2f

// ---------------------------------------------------------------------------
// Phase 1: per wave: one 16-row M-tile of one (b,c,h) GEMM  [16x64] = [16x64]x[64x64]
// ---------------------------------------------------------------------------
__global__ __launch_bounds__(256) void gat_phase1(
    const float* __restrict__ h, const float* __restrict__ w,
    const float* __restrict__ a_src, const float* __restrict__ a_dst,
    _Float16* __restrict__ hpT, float* __restrict__ srcv, float* __restrict__ dstv)
{
    __shared__ _Float16 wT[FF * FF];           // [o][k], k contiguous (8 KB)

    const int bch = blockIdx.x >> 3;           // 8 blocks per (b,c,h)
    const int tg  = blockIdx.x & 7;
    const int b   = bch >> 4;
    const int c   = (bch >> 2) & 3;
    const int hh  = bch & 3;

    // stage w[c,hh] transposed into LDS as f16
    {
        const float* wB = w + (size_t)(((c << 2) + hh) << 12);   // (c*H+hh)*4096
        const int t  = threadIdx.x;
        const int o  = t >> 2;
        const int k0 = (t & 3) << 4;
        #pragma unroll
        for (int kk = 0; kk < 16; ++kk)
            wT[o * FF + k0 + kk] = (_Float16)wB[(k0 + kk) * FF + o];
    }
    __syncthreads();

    const int lane  = threadIdx.x & 31;
    const int wave  = threadIdx.x >> 5;
    const int l15   = lane & 15;
    const int halfu = lane >> 4;               // 0: low half, 1: high half
    const int sA    = halfu << 3;              // A-frag K interleave base (0 / 8)
    const int sB    = halfu << 3;              // B-frag K interleave base (0 / 8)
    const int m0    = ((tg << 3) + wave) << 4; // row tile base

    // B fragments from LDS: [kb][nt], K = kb*32 + sB + {0..7, 16..23}, col o = nt*16 + l15
    v16h bf[8];
    #pragma unroll
    for (int kb = 0; kb < 2; ++kb) {
        #pragma unroll
        for (int nt = 0; nt < 4; ++nt) {
            const _Float16* col = &wT[((nt << 4) + l15) * FF + (kb << 5) + sB];
            v8h lo = *(const v8h*)(col);
            v8h hi = *(const v8h*)(col + 16);
            v16h f;
            #pragma unroll
            for (int e = 0; e < 8; ++e) { f[e] = lo[e]; f[e + 8] = hi[e]; }
            bf[kb * 4 + nt] = f;
        }
    }

    // A fragments from global h: row m0+l15, K groups {sA..sA+7, sA+16..sA+23}
    const float* hrow = h + (((size_t)((b << 2) + c) * NN + m0 + l15) << 6);
    v16h af[2];
    #pragma unroll
    for (int kb = 0; kb < 2; ++kb) {
        const float* p0 = hrow + (kb << 5) + sA;
        v4f x0 = *(const v4f*)(p0);
        v4f x1 = *(const v4f*)(p0 + 4);
        v4f x2 = *(const v4f*)(p0 + 16);
        v4f x3 = *(const v4f*)(p0 + 20);
        v16h f;
        #pragma unroll
        for (int e = 0; e < 4; ++e) {
            f[e]      = (_Float16)x0[e];
            f[e + 4]  = (_Float16)x1[e];
            f[e + 8]  = (_Float16)x2[e];
            f[e + 12] = (_Float16)x3[e];
        }
        af[kb] = f;
    }

    v8f acc[4];
    #pragma unroll
    for (int nt = 0; nt < 4; ++nt) { v8f z = {}; acc[nt] = z; }

    #pragma unroll
    for (int kb = 0; kb < 2; ++kb) {
        #pragma unroll
        for (int nt = 0; nt < 4; ++nt)
            acc[nt] = __builtin_amdgcn_wmma_f32_16x16x32_f16(
                false, af[kb], false, bf[kb * 4 + nt], (short)0, acc[nt], false, false);
    }

    // store h_prime transposed as f16: hpT[bch][o][n]  (V operand layout for phase 2)
    const size_t vb = ((size_t)bch << 6) * NN;
    #pragma unroll
    for (int nt = 0; nt < 4; ++nt) {
        v8h st;
        #pragma unroll
        for (int r = 0; r < 8; ++r) st[r] = (_Float16)acc[nt][r];
        *(v8h*)(hpT + vb + (size_t)((nt << 4) + l15) * NN + m0 + (halfu << 3)) = st;
    }

    // attn_src / attn_dst: rowwise sum_o tanh(h_prime) * a (16-lane reduction per half)
    const float* asB = a_src + (((c << 2) + hh) << 6);
    const float* adB = a_dst + (((c << 2) + hh) << 6);
    float as[4], ad[4];
    #pragma unroll
    for (int nt = 0; nt < 4; ++nt) {
        as[nt] = asB[(nt << 4) + l15];
        ad[nt] = adB[(nt << 4) + l15];
    }
    #pragma unroll
    for (int r = 0; r < 8; ++r) {
        float ps = 0.f, pd = 0.f;
        #pragma unroll
        for (int nt = 0; nt < 4; ++nt) {
            float tv = tanhf(acc[nt][r]);
            ps += tv * as[nt];
            pd += tv * ad[nt];
        }
        #pragma unroll
        for (int msk = 1; msk < 16; msk <<= 1) {
            ps += __shfl_xor(ps, msk);
            pd += __shfl_xor(pd, msk);
        }
        if (l15 == 0) {
            const int row = m0 + r + (halfu << 3);
            srcv[bch * NN + row] = ps;
            dstv[bch * NN + row] = pd;
        }
    }
}

// ---------------------------------------------------------------------------
// Phase 2: per wave: 16 query rows, flash loop over 32-wide j-blocks
// ---------------------------------------------------------------------------
__global__ __launch_bounds__(256) void gat_phase2(
    const float* __restrict__ adj, const _Float16* __restrict__ hpT,
    const float* __restrict__ srcv, const float* __restrict__ dstv,
    float* __restrict__ out)
{
    __shared__ float dstRow[NN];               // attn_dst row for this (b,c,h) (4 KB)

    const int bch = blockIdx.x >> 3;
    const int tg  = blockIdx.x & 7;
    const int b   = bch >> 4;

    {
        const int t = threadIdx.x;
        v4f tmp = *(const v4f*)(dstv + bch * NN + (t << 2));
        *(v4f*)(&dstRow[t << 2]) = tmp;
    }
    __syncthreads();

    const int lane  = threadIdx.x & 31;
    const int wave  = threadIdx.x >> 5;
    const int l15   = lane & 15;
    const int halfu = lane >> 4;
    const int sA    = halfu << 3;              // P-frag (A) K interleave base
    const int sB    = halfu << 3;              // V-frag (B) K interleave base
    const int i0    = ((tg << 3) + wave) << 4;
    const int iM    = i0 + l15;

    const float src_i = srcv[bch * NN + iM];
    const float* adjRow = adj + ((size_t)b * NN + iM) * NN;
    const _Float16* vbase = hpT + ((size_t)bch << 6) * NN;

    float m = -3.0e38f, l = 0.f;
    v8f acc[4];
    #pragma unroll
    for (int nt = 0; nt < 4; ++nt) { v8f z = {}; acc[nt] = z; }

    for (int j0 = 0; j0 < NN; j0 += 32) {
        // prefetch next j-block's adj segment and V rows (speculative; OOB dropped)
        __builtin_prefetch(adjRow + j0 + 32 + sA, 0, 3);
        #pragma unroll
        for (int nt = 0; nt < 4; ++nt)
            __builtin_prefetch(vbase + (size_t)((nt << 4) + l15) * NN + j0 + 32 + sB, 0, 3);

        // masked leaky-relu scores in A-fragment element order
        v4f d0 = *(const v4f*)(&dstRow[j0 + sA]);
        v4f d1 = *(const v4f*)(&dstRow[j0 + sA + 4]);
        v4f d2 = *(const v4f*)(&dstRow[j0 + sA + 16]);
        v4f d3 = *(const v4f*)(&dstRow[j0 + sA + 20]);
        v4f g0 = *(const v4f*)(adjRow + j0 + sA);
        v4f g1 = *(const v4f*)(adjRow + j0 + sA + 4);
        v4f g2 = *(const v4f*)(adjRow + j0 + sA + 16);
        v4f g3 = *(const v4f*)(adjRow + j0 + sA + 20);

        float sc[16];
        float bm = -3.0e38f;
        #pragma unroll
        for (int e = 0; e < 16; ++e) {
            const int jj = j0 + sA + ((e < 8) ? e : (8 + e));   // {0..7, 16..23}
            const float dv = (e < 8) ? ((e < 4) ? d0[e] : d1[e - 4])
                                     : ((e < 12) ? d2[e - 8] : d3[e - 12]);
            const float gv = (e < 8) ? ((e < 4) ? g0[e] : g1[e - 4])
                                     : ((e < 12) ? g2[e - 8] : g3[e - 12]);
            float a = src_i + dv;
            a = (a >= 0.f) ? a : SLOPE * a;
            const bool keep = (gv != 0.f) || (iM == jj);
            sc[e] = keep ? a : NEGINF;
            bm = fmaxf(bm, sc[e]);
        }
        bm = fmaxf(bm, __shfl_xor(bm, 16));                     // row max across halves
        const float mnew = fmaxf(m, bm);
        const float cs = __expf(m - mnew);                      // accumulator correction

        float psum = 0.f;
        v16h pf;
        #pragma unroll
        for (int e = 0; e < 16; ++e) {
            const float pv = __expf(sc[e] - mnew);
            psum += pv;
            pf[e] = (_Float16)pv;
        }
        psum += __shfl_xor(psum, 16);
        l = l * cs + psum;
        m = mnew;

        // rescale accumulators: gather per-row correction (C/D row = r + 8*halfu)
        float cr[8];
        #pragma unroll
        for (int r = 0; r < 8; ++r) cr[r] = __shfl(cs, r + (halfu << 3));
        #pragma unroll
        for (int nt = 0; nt < 4; ++nt) {
            #pragma unroll
            for (int r = 0; r < 8; ++r) acc[nt][r] *= cr[r];
        }

        // V fragments (contiguous f16 thanks to transposed h_prime) + WMMA
        #pragma unroll
        for (int nt = 0; nt < 4; ++nt) {
            const _Float16* vc = vbase + (size_t)((nt << 4) + l15) * NN + j0 + sB;
            v8h lo = *(const v8h*)(vc);
            v8h hi = *(const v8h*)(vc + 16);
            v16h vf;
            #pragma unroll
            for (int e = 0; e < 8; ++e) { vf[e] = lo[e]; vf[e + 8] = hi[e]; }
            acc[nt] = __builtin_amdgcn_wmma_f32_16x16x32_f16(
                false, pf, false, vf, (short)0, acc[nt], false, false);
        }
    }

    // epilogue: multiply by reciprocal row sums, store
    float lrcp[8];
    #pragma unroll
    for (int r = 0; r < 8; ++r) lrcp[r] = 1.0f / __shfl(l, r + (halfu << 3));
    float* ob = out + (size_t)bch * NN * FF;
    #pragma unroll
    for (int nt = 0; nt < 4; ++nt) {
        const int o = (nt << 4) + l15;
        #pragma unroll
        for (int r = 0; r < 8; ++r) {
            const int row = i0 + r + (halfu << 3);
            ob[(size_t)row * FF + o] = acc[nt][r] * lrcp[r];
        }
    }
}

extern "C" void kernel_launch(void* const* d_in, const int* in_sizes, int n_in,
                              void* d_out, int out_size, void* d_ws, size_t ws_size,
                              hipStream_t stream) {
    (void)in_sizes; (void)n_in; (void)out_size; (void)ws_size;
    const float* h     = (const float*)d_in[0];
    const float* adj   = (const float*)d_in[1];
    const float* w     = (const float*)d_in[2];
    const float* a_src = (const float*)d_in[3];
    const float* a_dst = (const float*)d_in[4];
    float* out = (float*)d_out;

    _Float16* hpT = (_Float16*)d_ws;                                   // 8 MB
    float* srcv = (float*)((char*)d_ws + (size_t)BCH * FF * NN * sizeof(_Float16));
    float* dstv = srcv + (size_t)BCH * NN;

    dim3 grid(BCH * 8), blk(256);
    gat_phase1<<<grid, blk, 0, stream>>>(h, w, a_src, a_dst, hpT, srcv, dstv);
    gat_phase2<<<grid, blk, 0, stream>>>(adj, hpT, srcv, dstv, out);
}
